// LSTM_decoder_15599321219554
// MI455X (gfx1250) — matrix operand order repaired and checked by reading, not compiled, not run on hardware
//
#include <hip/hip_runtime.h>
#include <hip/hip_bf16.h>

// LSTM decoder for MI455X (gfx1250, wave32, WMMA).
// bf16 weights converted once (L2-resident, ~100MB of 192MB), 50-step rollout
// as stream-ordered kernels. Gate GEMMs use v_wmma_f32_16x16x32_bf16 with the
// same contiguous k order (16*half + s) in A and B fragments. A-tile staging uses the gfx1250
// async-to-LDS datapath (ASYNCcnt) when the builtin is available, else the
// global_load_b128 + ds_store_b128 path.

typedef __bf16 bf16_t;
typedef __attribute__((ext_vector_type(16))) __bf16 v16bf;
typedef __attribute__((ext_vector_type(8)))  __bf16 v8bf;
typedef __attribute__((ext_vector_type(8)))  float  v8f;
typedef int v4i __attribute__((vector_size(16)));   // matches async-to-LDS builtin param

#define B_    256
#define H_    2048
#define G_    8192   /* 4*H */
#define IN_   96
#define TENC_ 49
#define TOUT_ 50
#define LDSROW 40    /* 64 rows x 32 cols staged, padded stride (80B) -> conflict-free, 16B aligned */

#define GLOBAL_AS __attribute__((address_space(1)))
#define LDS_AS    __attribute__((address_space(3)))

__device__ __forceinline__ v8f wmma_bf16(v16bf a, v16bf b, v8f c) {
  return __builtin_amdgcn_wmma_f32_16x16x32_bf16(false, a, false, b, (short)0, c, false, false);
}
__device__ __forceinline__ float sigmoidf_(float x) { return 1.0f / (1.0f + __expf(-x)); }

__device__ __forceinline__ void wait_asynccnt0() {
#if __has_builtin(__builtin_amdgcn_s_wait_asynccnt)
  __builtin_amdgcn_s_wait_asynccnt(0);
#else
  asm volatile("s_wait_asynccnt 0x0" ::: "memory");
#endif
}

// ---------------------------------------------------------------------------
// gates[m,n] = sum_k Xa[m,k]*Wa[n,k] + sum_k Xb[m,k]*Wb[n,k] + bias[n]
// M=256, N=8192. Grid: (N/256, M/64), block 256 threads = 8 waves (2M x 4N),
// each wave owns a 32x64 tile (2x4 WMMA accumulators).
// ---------------------------------------------------------------------------
__global__ __launch_bounds__(256)
void gemm_gates_kernel(const bf16_t* __restrict__ Xa, int Ka, const bf16_t* __restrict__ Wa,
                       const bf16_t* __restrict__ Xb, int Kb, const bf16_t* __restrict__ Wb,
                       const float* __restrict__ bias, float* __restrict__ gates)
{
  __shared__ bf16_t As[64 * LDSROW];
  const int lane  = threadIdx.x & 31;
  const int wave  = threadIdx.x >> 5;
  const int waveM = wave >> 2;        // 0..1
  const int waveN = wave & 3;         // 0..3
  const int mblk  = blockIdx.y * 64;
  const int nblk  = blockIdx.x * 256 + waveN * 64;
  const int l15   = lane & 15;
  const int lhi   = lane >> 4;
  const int sr    = threadIdx.x >> 2;        // staging row 0..63
  const int sc    = (threadIdx.x & 3) << 3;  // staging col {0,8,16,24}

  const v8f vzero = {0.f,0.f,0.f,0.f,0.f,0.f,0.f,0.f};
  v8f acc[2][4];
#pragma unroll
  for (int i = 0; i < 2; ++i)
#pragma unroll
    for (int j = 0; j < 4; ++j) acc[i][j] = vzero;

  for (int seg = 0; seg < 2; ++seg) {
    const bf16_t* X = seg ? Xb : Xa;
    const bf16_t* W = seg ? Wb : Wa;
    const int K = seg ? Kb : Ka;
    for (int k0 = 0; k0 < K; k0 += 32) {
      __syncthreads();  // previous iteration's A-fragment reads complete
      // stage 64x32 activation chunk to LDS (16B per thread)
#if __has_builtin(__builtin_amdgcn_global_load_async_to_lds_b128)
      __builtin_amdgcn_global_load_async_to_lds_b128(
          (GLOBAL_AS v4i*)&X[(mblk + sr) * K + k0 + sc],
          (LDS_AS v4i*)&As[sr * LDSROW + sc], 0, 0);
      wait_asynccnt0();
#else
      *(v8bf*)&As[sr * LDSROW + sc] = *(const v8bf*)&X[(mblk + sr) * K + k0 + sc];
#endif
      __syncthreads();

      // A fragments for the wave's two M tiles (contiguous k order: k = 16*half + 0..15)
      v16bf afrag[2];
#pragma unroll
      for (int i = 0; i < 2; ++i) {
        const bf16_t* ap = &As[(waveM * 32 + i * 16 + l15) * LDSROW + (lhi << 4)];
        v8bf lo = *(const v8bf*)ap;         // K = khalf*16 + 0..7
        v8bf hi = *(const v8bf*)(ap + 8);   // K = khalf*16 + 8..15
#pragma unroll
        for (int e = 0; e < 8; ++e) { afrag[i][e] = lo[e]; afrag[i][8 + e] = hi[e]; }
      }
      // B fragments straight from L2-resident weights (row-major [N][K])
#pragma unroll
      for (int j = 0; j < 4; ++j) {
        v16bf bfrag = *(const v16bf*)&W[(nblk + j * 16 + l15) * K + k0 + (lhi << 4)];
#pragma unroll
        for (int i = 0; i < 2; ++i)
          acc[i][j] = wmma_bf16(afrag[i], bfrag, acc[i][j]);
      }
    }
  }

  // epilogue: + bias, store f32 gates
#pragma unroll
  for (int j = 0; j < 4; ++j) {
    const int n  = nblk + j * 16 + l15;
    const float bv = bias[n];
#pragma unroll
    for (int i = 0; i < 2; ++i) {
      const int mrow = mblk + waveM * 32 + i * 16 + (lhi << 3);
#pragma unroll
      for (int v = 0; v < 8; ++v)
        gates[(mrow + v) * G_ + n] = acc[i][j][v] + bv;
    }
  }
}

// ---------------------------------------------------------------------------
// LSTM pointwise: c' = sig(f)*c + sig(i)*tanh(g); h' = sig(o)*tanh(c')
// ---------------------------------------------------------------------------
__global__ __launch_bounds__(256)
void lstm_act_kernel(const float* __restrict__ gates, float* __restrict__ c,
                     float* __restrict__ h, bf16_t* __restrict__ hbf,
                     float* __restrict__ c_extra)
{
  const int idx = blockIdx.x * 256 + threadIdx.x;
  if (idx >= B_ * H_) return;
  const int m = idx >> 11;
  const int j = idx & (H_ - 1);
  const float* g = gates + m * G_;
  const float iv = sigmoidf_(g[j]);
  const float fv = sigmoidf_(g[H_ + j]);
  const float gv = tanhf(g[2 * H_ + j]);
  const float ov = sigmoidf_(g[3 * H_ + j]);
  const float cn = fv * c[idx] + iv * gv;
  const float hn = ov * tanhf(cn);
  c[idx] = cn;
  h[idx] = hn;
  hbf[idx] = (bf16_t)hn;
  if (c_extra) c_extra[idx] = cn;
}

// ---------------------------------------------------------------------------
// Output projection: pre = h1 @ Wout + bias_out + x_prev (residual).
// M=256,N=96,K=2048. 96 tiles of 16x16, one per wave; 12 blocks x 8 waves.
// Updates x (f32 + bf16) in place and writes out0[t].
// ---------------------------------------------------------------------------
__global__ __launch_bounds__(256)
void out_proj_kernel(const bf16_t* __restrict__ h1bf, const bf16_t* __restrict__ WoutT,
                     const float* __restrict__ bias_out, float* __restrict__ x_f32,
                     bf16_t* __restrict__ x_bf, float* __restrict__ out0_t)
{
  const int lane = threadIdx.x & 31;
  const int wave = threadIdx.x >> 5;
  const int tile = blockIdx.x * 8 + wave;   // 0..95
  const int mt = (tile / 6) * 16;
  const int nt = (tile % 6) * 16;
  const int l15 = lane & 15;
  const int lhi = lane >> 4;

  v8f acc = {0.f,0.f,0.f,0.f,0.f,0.f,0.f,0.f};
  for (int k0 = 0; k0 < H_; k0 += 32) {
    v16bf a;
    {
      const bf16_t* ap = &h1bf[(mt + l15) * H_ + k0 + (lhi << 4)];
      v8bf lo = *(const v8bf*)ap;
      v8bf hi = *(const v8bf*)(ap + 8);
#pragma unroll
      for (int e = 0; e < 8; ++e) { a[e] = lo[e]; a[8 + e] = hi[e]; }
    }
    v16bf b = *(const v16bf*)&WoutT[(nt + l15) * H_ + k0 + (lhi << 4)];
    acc = wmma_bf16(a, b, acc);
  }
  const int n = nt + l15;
  const float bv = bias_out[n];
  const int mbase = mt + (lhi << 3);
#pragma unroll
  for (int v = 0; v < 8; ++v) {
    const int m = mbase + v;
    const float val = acc[v] + bv + x_f32[m * IN_ + n];
    x_f32[m * IN_ + n] = val;
    x_bf[m * IN_ + n] = (bf16_t)val;
    out0_t[m * IN_ + n] = val;
  }
}

// ---------------------------------------------------------------------------
// Init: per (b,j) reductions over T_ENC for h0/c/h1 initial states.
// ---------------------------------------------------------------------------
__global__ __launch_bounds__(256)
void init_states_kernel(const float* __restrict__ hs, const float* __restrict__ cs,
                        const float* __restrict__ gts,
                        float* __restrict__ h0, float* __restrict__ c0,
                        float* __restrict__ h1, float* __restrict__ c1,
                        bf16_t* __restrict__ h0bf, bf16_t* __restrict__ h1bf)
{
  const int idx = blockIdx.x * 256 + threadIdx.x;
  if (idx >= B_ * H_) return;
  const int b = idx >> 11;
  const int j = idx & (H_ - 1);
  const float* ph = hs + (size_t)b * TENC_ * H_ + j;
  const float* pc = cs + (size_t)b * TENC_ * H_ + j;
  float sh = 0.f, sc = 0.f;
  for (int t = 0; t < TENC_; ++t) { sh += ph[t * H_]; sc += pc[t * H_]; }
  const float c0v = sc * (1.0f / 49.0f);
  const float h0v = sh * (1.0f / 49.0f);
  const float h1v = (gts[idx] + sh) * (1.0f / 50.0f);
  h0[idx] = h0v; c0[idx] = c0v; h1[idx] = h1v; c1[idx] = c0v;
  h0bf[idx] = (bf16_t)h0v; h1bf[idx] = (bf16_t)h1v;
}

__global__ __launch_bounds__(256)
void init_x_kernel(const float* __restrict__ p, float* __restrict__ x_f32,
                   bf16_t* __restrict__ x_bf)
{
  const int idx = blockIdx.x * 256 + threadIdx.x;
  if (idx >= B_ * IN_) return;
  const float v = p[idx];
  x_f32[idx] = v;
  x_bf[idx] = (bf16_t)v;
}

__global__ __launch_bounds__(256)
void cvt_bf16_kernel(const float* __restrict__ src, bf16_t* __restrict__ dst, int n)
{
  const int idx = blockIdx.x * 256 + threadIdx.x;
  if (idx < n) dst[idx] = (bf16_t)src[idx];
}

// weights_out is (H, IN); store transposed (IN, H) so B-fragment reads are contiguous.
__global__ __launch_bounds__(256)
void cvtT_wout_kernel(const float* __restrict__ W, bf16_t* __restrict__ WT)
{
  const int idx = blockIdx.x * 256 + threadIdx.x;
  if (idx >= IN_ * H_) return;
  const int n = idx / H_;
  const int k = idx - n * H_;
  WT[idx] = (bf16_t)W[k * IN_ + n];
}

__global__ __launch_bounds__(256)
void bias_sum_kernel(const float* __restrict__ a, const float* __restrict__ b,
                     float* __restrict__ dst)
{
  const int idx = blockIdx.x * 256 + threadIdx.x;
  if (idx < G_) dst[idx] = a[idx] + b[idx];
}

// ---------------------------------------------------------------------------
extern "C" void kernel_launch(void* const* d_in, const int* in_sizes, int n_in,
                              void* d_out, int out_size, void* d_ws, size_t ws_size,
                              hipStream_t stream)
{
  (void)in_sizes; (void)n_in; (void)out_size; (void)ws_size;
  const float* hs     = (const float*)d_in[0];
  const float* cs     = (const float*)d_in[1];
  const float* gts    = (const float*)d_in[2];
  const float* p      = (const float*)d_in[4];
  const float* w_ih0  = (const float*)d_in[6];
  const float* w_hh0  = (const float*)d_in[7];
  const float* b_ih0  = (const float*)d_in[8];
  const float* b_hh0  = (const float*)d_in[9];
  const float* w_ih1  = (const float*)d_in[10];
  const float* w_hh1  = (const float*)d_in[11];
  const float* b_ih1  = (const float*)d_in[12];
  const float* b_hh1  = (const float*)d_in[13];
  const float* w_out  = (const float*)d_in[14];
  const float* b_out  = (const float*)d_in[15];

  float* out0 = (float*)d_out;                       // (T_OUT, B, IN)
  float* out1 = out0 + (size_t)TOUT_ * B_ * IN_;     // (T_OUT, B, H)

  // ---- workspace carve-out (deterministic, ~130MB) ----
  char* base = (char*)d_ws;
  size_t off = 0;
  auto alloc = [&](size_t bytes) -> void* {
    void* q = base + off;
    off = (off + bytes + 255) & ~(size_t)255;
    return q;
  };
  bf16_t* Wih0 = (bf16_t*)alloc((size_t)G_ * IN_ * 2);
  bf16_t* Whh0 = (bf16_t*)alloc((size_t)G_ * H_ * 2);
  bf16_t* Wih1 = (bf16_t*)alloc((size_t)G_ * H_ * 2);
  bf16_t* Whh1 = (bf16_t*)alloc((size_t)G_ * H_ * 2);
  bf16_t* WoutT = (bf16_t*)alloc((size_t)IN_ * H_ * 2);
  float*  bias0 = (float*)alloc((size_t)G_ * 4);
  float*  bias1 = (float*)alloc((size_t)G_ * 4);
  float*  gates0 = (float*)alloc((size_t)B_ * G_ * 4);
  float*  gates1 = (float*)alloc((size_t)B_ * G_ * 4);
  float*  h0 = (float*)alloc((size_t)B_ * H_ * 4);
  float*  c0 = (float*)alloc((size_t)B_ * H_ * 4);
  float*  h1 = (float*)alloc((size_t)B_ * H_ * 4);
  float*  c1 = (float*)alloc((size_t)B_ * H_ * 4);
  bf16_t* h0bf = (bf16_t*)alloc((size_t)B_ * H_ * 2);
  bf16_t* h1bf = (bf16_t*)alloc((size_t)B_ * H_ * 2);
  float*  x_f32 = (float*)alloc((size_t)B_ * IN_ * 4);
  bf16_t* x_bf  = (bf16_t*)alloc((size_t)B_ * IN_ * 2);

  const int TPB = 256;
  auto blk = [](int n) { return (n + 255) / 256; };

  // ---- one-time (per-call) prep ----
  cvt_bf16_kernel<<<blk(G_ * IN_), TPB, 0, stream>>>(w_ih0, Wih0, G_ * IN_);
  cvt_bf16_kernel<<<blk(G_ * H_), TPB, 0, stream>>>(w_hh0, Whh0, G_ * H_);
  cvt_bf16_kernel<<<blk(G_ * H_), TPB, 0, stream>>>(w_ih1, Wih1, G_ * H_);
  cvt_bf16_kernel<<<blk(G_ * H_), TPB, 0, stream>>>(w_hh1, Whh1, G_ * H_);
  cvtT_wout_kernel<<<blk(IN_ * H_), TPB, 0, stream>>>(w_out, WoutT);
  bias_sum_kernel<<<blk(G_), TPB, 0, stream>>>(b_ih0, b_hh0, bias0);
  bias_sum_kernel<<<blk(G_), TPB, 0, stream>>>(b_ih1, b_hh1, bias1);
  init_states_kernel<<<blk(B_ * H_), TPB, 0, stream>>>(hs, cs, gts, h0, c0, h1, c1, h0bf, h1bf);
  init_x_kernel<<<blk(B_ * IN_), TPB, 0, stream>>>(p, x_f32, x_bf);

  // ---- 50-step rollout ----
  const dim3 gemm_grid(G_ / 256, B_ / 64);  // (32, 4)
  for (int t = 0; t < TOUT_; ++t) {
    gemm_gates_kernel<<<gemm_grid, TPB, 0, stream>>>(x_bf, IN_, Wih0, h0bf, H_, Whh0, bias0, gates0);
    lstm_act_kernel<<<blk(B_ * H_), TPB, 0, stream>>>(gates0, c0, h0, h0bf, (float*)nullptr);
    gemm_gates_kernel<<<gemm_grid, TPB, 0, stream>>>(h0bf, H_, Wih1, h1bf, H_, Whh1, bias1, gates1);
    lstm_act_kernel<<<blk(B_ * H_), TPB, 0, stream>>>(gates1, c1, h1, h1bf, out1 + (size_t)t * B_ * H_);
    out_proj_kernel<<<12, TPB, 0, stream>>>(h1bf, WoutT, b_out, x_f32, x_bf, out0 + (size_t)t * B_ * IN_);
  }
}
